// TemporalAttention_31267361914951
// MI455X (gfx1250) — matrix operand order repaired and checked by reading, not compiled, not run on hardware
//
#include <hip/hip_runtime.h>
#include <hip/hip_bf16.h>
#include <stdint.h>

typedef __attribute__((ext_vector_type(16))) _Float16 v16h;
typedef __attribute__((ext_vector_type(8)))  float    v8f;

union Frag16 {
    v16h     v;
    _Float16 h[16];
    uint4    q[2];
};

// ---- WMMA helper: D = A(16x32 f16) * B(32x16 f16) + C(16x16 f32) ----
__device__ __forceinline__ v8f wmma_f16(const Frag16& a, const Frag16& b, v8f c) {
    return __builtin_amdgcn_wmma_f32_16x16x32_f16(false, a.v, false, b.v,
                                                  (short)0, c, false, false);
}

// A-matrix fragment (16x32, row = lane&15). half=lane>>4.
// half 0: K = kbase+[0..7] and kbase+[16..23]; half 1: +8 on each chunk.
__device__ __forceinline__ void load_a(Frag16& f, const _Float16* p, int half) {
    f.q[0] = *(const uint4*)(p + 8 * half);
    f.q[1] = *(const uint4*)(p + 16 + 8 * half);
}

// B-matrix fragment (32x16, col = lane&15). half h holds K = kbase+8h+[0..7] and kbase+16+8h+[0..7].
__device__ __forceinline__ void load_b(Frag16& f, const _Float16* p, int half) {
    f.q[0] = *(const uint4*)(p + 8 * half);
    f.q[1] = *(const uint4*)(p + 16 + 8 * half);
}

// =====================================================================
// Kernel 1: f32 -> f16 transpose convert  (dst[n*K + k] = src[k*N + n])
// =====================================================================
__global__ void k_transpose_f16(const float* __restrict__ src,
                                _Float16* __restrict__ dst, int K, int N) {
    int id = blockIdx.x * blockDim.x + threadIdx.x;
    if (id >= K * N) return;
    int k = id / N, n = id % N;
    dst[(size_t)n * K + k] = (_Float16)src[id];
}

// =====================================================================
// Kernel 2: LayerNorm rows of 512 -> f16   (one 256-thread block per row)
// =====================================================================
__global__ void k_layernorm(const float* __restrict__ x,
                            const float* __restrict__ gamma,
                            const float* __restrict__ beta,
                            _Float16* __restrict__ xn) {
    const int D = 512;
    int row = blockIdx.x, tid = threadIdx.x;
    const float* xr = x + (size_t)row * D;
    float v0 = xr[tid], v1 = xr[tid + 256];

    __shared__ float rs[256], rq[256];
    rs[tid] = v0 + v1;
    rq[tid] = v0 * v0 + v1 * v1;
    __syncthreads();
    for (int off = 128; off > 0; off >>= 1) {
        if (tid < off) { rs[tid] += rs[tid + off]; rq[tid] += rq[tid + off]; }
        __syncthreads();
    }
    float mu   = rs[0] * (1.0f / D);
    float var  = rq[0] * (1.0f / D) - mu * mu;
    float rstd = rsqrtf(var + 1e-5f);

    _Float16* o = xn + (size_t)row * D;
    o[tid]       = (_Float16)((v0 - mu) * rstd * gamma[tid] + beta[tid]);
    o[tid + 256] = (_Float16)((v1 - mu) * rstd * gamma[tid + 256] + beta[tid + 256]);
}

// =====================================================================
// Kernel 3: QKV projection. One wave -> 16x64 tile of (8192 x 1536).
// Software-pipelined k-loop (double-buffered fragments).
// Writes q (pre-scaled 1/8) [bh,t,d], k [bh,t,d], v transposed [bh,d,t].
// =====================================================================
__global__ void __launch_bounds__(256)
k_qkv_gemm(const _Float16* __restrict__ xn, const _Float16* __restrict__ wT,
           _Float16* __restrict__ qb, _Float16* __restrict__ kb,
           _Float16* __restrict__ vT) {
    const int D = 512, T = 4096, H = 8, NT = 24;  // 1536/64
    int wid  = blockIdx.x * 8 + (threadIdx.x >> 5);
    int lane = threadIdx.x & 31, half = lane >> 4, l16 = lane & 15;
    int mt = wid / NT, nt = wid % NT;

    const _Float16* aptr = xn + (size_t)(mt * 16 + l16) * D;
    const _Float16* bptr[4];
#pragma unroll
    for (int s = 0; s < 4; ++s)
        bptr[s] = wT + (size_t)(nt * 64 + s * 16 + l16) * D;

    Frag16 aC, aN, bC[4], bN[4];
    load_a(aC, aptr, half);
#pragma unroll
    for (int s = 0; s < 4; ++s) load_b(bC[s], bptr[s], half);

    v8f acc[4] = {};
#pragma unroll
    for (int kk = 0; kk < D; kk += 32) {
        if (kk + 32 < D) {
            load_a(aN, aptr + kk + 32, half);
#pragma unroll
            for (int s = 0; s < 4; ++s) load_b(bN[s], bptr[s] + kk + 32, half);
        }
#pragma unroll
        for (int s = 0; s < 4; ++s) acc[s] = wmma_f16(aC, bC[s], acc[s]);
        if (kk + 32 < D) {
            aC = aN;
#pragma unroll
            for (int s = 0; s < 4; ++s) bC[s] = bN[s];
        }
    }

#pragma unroll
    for (int s = 0; s < 4; ++s) {
        int c = nt * 64 + s * 16 + l16;
        int which = c >> 9, rem = c & 511;
        int h = rem >> 6, d = rem & 63;
#pragma unroll
        for (int i = 0; i < 8; ++i) {
            int m = mt * 16 + i + 8 * half;
            int b_ = m >> 12, t = m & 4095;
            size_t bh = (size_t)(b_ * H + h);
            float val = acc[s][i];
            if (which == 0)      qb[(bh * T + t) * 64 + d] = (_Float16)(val * 0.125f);
            else if (which == 1) kb[(bh * T + t) * 64 + d] = (_Float16)val;
            else                 vT[(bh * 64 + d) * T + t] = (_Float16)val;
        }
    }
}

// =====================================================================
// Kernel 4: causal flash attention. One wave per (bh, 16-row Q tile).
// V fragments hoisted before softmax; P staged through per-wave LDS.
// =====================================================================
__global__ void __launch_bounds__(256)
k_attention(const _Float16* __restrict__ qm, const _Float16* __restrict__ km,
            const _Float16* __restrict__ vT, _Float16* __restrict__ om) {
    const int T = 4096, HD = 64;
    __shared__ __align__(16) _Float16 plds[8 * 16 * 32];

    int warp = threadIdx.x >> 5, lane = threadIdx.x & 31;
    int half = lane >> 4, l16 = lane & 15;
    int wid = blockIdx.x * 8 + warp;     // 0..4095
    int bh = wid >> 8;                   // 0..15  (b*8 + h)
    int qbase = (wid & 255) * 16;
    _Float16* pbuf = plds + warp * (16 * 32);

    const _Float16* qrow = qm + ((size_t)bh * T + (qbase + l16)) * HD;
    Frag16 qa0, qa1;
    load_a(qa0, qrow, half);
    load_a(qa1, qrow + 32, half);

    float m_run[8], l_run[8];
    v8f o_acc[4] = {};
#pragma unroll
    for (int i = 0; i < 8; ++i) { m_run[i] = -3.0e38f; l_run[i] = 0.0f; }

    const int smax = qbase + 15;
    for (int s0 = 0; s0 <= smax; s0 += 32) {
        // K fragments
        const _Float16* krow0 = km + ((size_t)bh * T + (s0 + l16)) * HD;
        const _Float16* krow1 = krow0 + 16 * HD;
        Frag16 kf0, kf1, kf2, kf3;
        load_b(kf0, krow0, half);
        load_b(kf1, krow0 + 32, half);
        load_b(kf2, krow1, half);
        load_b(kf3, krow1 + 32, half);
        // V fragments early: independent of softmax, latency hidden behind it
        Frag16 vb[4];
#pragma unroll
        for (int nc = 0; nc < 4; ++nc)
            load_b(vb[nc], vT + ((size_t)bh * HD + nc * 16 + l16) * T + s0, half);

        v8f sc0 = {}, sc1 = {};
        sc0 = wmma_f16(qa0, kf0, sc0);
        sc0 = wmma_f16(qa1, kf1, sc0);
        sc1 = wmma_f16(qa0, kf2, sc1);
        sc1 = wmma_f16(qa1, kf3, sc1);

        int key0 = s0 + l16, key1 = key0 + 16;
        float p0[8], p1[8], rmax[8], rsum[8];
#pragma unroll
        for (int i = 0; i < 8; ++i) {
            int row = qbase + i + 8 * half;
            p0[i] = (key0 <= row) ? sc0[i] : -1.0e30f;
            p1[i] = (key1 <= row) ? sc1[i] : -1.0e30f;
            rmax[i] = fmaxf(p0[i], p1[i]);
        }
#pragma unroll
        for (int msk = 1; msk <= 8; msk <<= 1)
#pragma unroll
            for (int i = 0; i < 8; ++i)
                rmax[i] = fmaxf(rmax[i], __shfl_xor(rmax[i], msk, 32));
#pragma unroll
        for (int i = 0; i < 8; ++i) {
            float mnew  = fmaxf(m_run[i], rmax[i]);
            float alpha = __expf(m_run[i] - mnew);
            m_run[i] = mnew;
            p0[i] = __expf(p0[i] - mnew);
            p1[i] = __expf(p1[i] - mnew);
            rsum[i] = p0[i] + p1[i];
            l_run[i] *= alpha;
#pragma unroll
            for (int nc = 0; nc < 4; ++nc) o_acc[nc][i] *= alpha;
        }
#pragma unroll
        for (int msk = 1; msk <= 8; msk <<= 1)
#pragma unroll
            for (int i = 0; i < 8; ++i)
                rsum[i] += __shfl_xor(rsum[i], msk, 32);
#pragma unroll
        for (int i = 0; i < 8; ++i) l_run[i] += rsum[i];

        // P (C-layout) -> LDS (row-major 16x32) -> A-fragment
#pragma unroll
        for (int i = 0; i < 8; ++i) {
            int r = i + 8 * half;
            pbuf[r * 32 + l16]      = (_Float16)p0[i];
            pbuf[r * 32 + 16 + l16] = (_Float16)p1[i];
        }
        asm volatile("" ::: "memory");
        __builtin_amdgcn_wave_barrier();
        Frag16 pa;
        load_a(pa, pbuf + l16 * 32, half);
        asm volatile("" ::: "memory");
        __builtin_amdgcn_wave_barrier();

#pragma unroll
        for (int nc = 0; nc < 4; ++nc)
            o_acc[nc] = wmma_f16(pa, vb[nc], o_acc[nc]);
    }

    int b_ = bh >> 3, h = bh & 7;
#pragma unroll
    for (int nc = 0; nc < 4; ++nc) {
        int d = nc * 16 + l16;
#pragma unroll
        for (int i = 0; i < 8; ++i) {
            int row = qbase + i + 8 * half;
            float val = o_acc[nc][i] / l_run[i];
            om[((size_t)b_ * T + row) * 512 + h * 64 + d] = (_Float16)val;
        }
    }
}

// =====================================================================
// Kernel 5: out-projection (8192x512 @ 512x512) + bias + residual -> f32
// Software-pipelined k-loop.
// =====================================================================
__global__ void __launch_bounds__(256)
k_outproj(const _Float16* __restrict__ a, const _Float16* __restrict__ wT,
          const float* __restrict__ bias, const float* __restrict__ resid,
          float* __restrict__ out) {
    const int D = 512;
    int wid  = blockIdx.x * 8 + (threadIdx.x >> 5);
    int lane = threadIdx.x & 31, half = lane >> 4, l16 = lane & 15;
    int mt = wid >> 3, nt = wid & 7;

    const _Float16* aptr = a + (size_t)(mt * 16 + l16) * D;
    const _Float16* bptr[4];
#pragma unroll
    for (int s = 0; s < 4; ++s)
        bptr[s] = wT + (size_t)(nt * 64 + s * 16 + l16) * D;

    Frag16 aC, aN, bC[4], bN[4];
    load_a(aC, aptr, half);
#pragma unroll
    for (int s = 0; s < 4; ++s) load_b(bC[s], bptr[s], half);

    v8f acc[4] = {};
#pragma unroll
    for (int kk = 0; kk < D; kk += 32) {
        if (kk + 32 < D) {
            load_a(aN, aptr + kk + 32, half);
#pragma unroll
            for (int s = 0; s < 4; ++s) load_b(bN[s], bptr[s] + kk + 32, half);
        }
#pragma unroll
        for (int s = 0; s < 4; ++s) acc[s] = wmma_f16(aC, bC[s], acc[s]);
        if (kk + 32 < D) {
            aC = aN;
#pragma unroll
            for (int s = 0; s < 4; ++s) bC[s] = bN[s];
        }
    }

#pragma unroll
    for (int s = 0; s < 4; ++s) {
        int c = nt * 64 + s * 16 + l16;
        float bv = bias[c];
#pragma unroll
        for (int i = 0; i < 8; ++i) {
            int m = mt * 16 + i + 8 * half;
            out[(size_t)m * D + c] = acc[s][i] + bv + resid[(size_t)m * D + c];
        }
    }
}

// =====================================================================
extern "C" void kernel_launch(void* const* d_in, const int* in_sizes, int n_in,
                              void* d_out, int out_size, void* d_ws, size_t ws_size,
                              hipStream_t stream) {
    const float* x     = (const float*)d_in[0];
    const float* w_qkv = (const float*)d_in[1];
    const float* w_out = (const float*)d_in[2];
    const float* b_out = (const float*)d_in[3];
    const float* gamma = (const float*)d_in[4];
    const float* beta  = (const float*)d_in[5];
    float* out = (float*)d_out;

    const int B = 2, T = 4096, D = 512, H = 8;
    const int M = B * T;  // 8192

    char* ws = (char*)d_ws;
    size_t off = 0;
    auto alloc = [&](size_t bytes) -> void* {
        void* p = ws + off;
        off += (bytes + 255) & ~(size_t)255;
        return p;
    };
    _Float16* xn    = (_Float16*)alloc((size_t)M * D * 2);          // 8.4 MB
    _Float16* wqkvT = (_Float16*)alloc((size_t)D * 3 * D * 2);      // 1.6 MB
    _Float16* woutT = (_Float16*)alloc((size_t)D * D * 2);          // 0.5 MB
    _Float16* qb    = (_Float16*)alloc((size_t)B * H * T * 64 * 2); // 8.4 MB
    _Float16* kb    = (_Float16*)alloc((size_t)B * H * T * 64 * 2); // 8.4 MB
    _Float16* vT    = (_Float16*)alloc((size_t)B * H * 64 * T * 2); // 8.4 MB
    _Float16* ao    = (_Float16*)alloc((size_t)M * D * 2);          // 8.4 MB

    k_transpose_f16<<<(D * 3 * D + 255) / 256, 256, 0, stream>>>(w_qkv, wqkvT, D, 3 * D);
    k_transpose_f16<<<(D * D + 255) / 256, 256, 0, stream>>>(w_out, woutT, D, D);
    k_layernorm<<<M, 256, 0, stream>>>(x, gamma, beta, xn);
    k_qkv_gemm<<<(M / 16) * 24 / 8, 256, 0, stream>>>(xn, wqkvT, qb, kb, vT);   // 1536 blocks
    k_attention<<<(B * H * (T / 16)) / 8, 256, 0, stream>>>(qb, kb, vT, ao);    // 512 blocks
    k_outproj<<<(M / 16) * 8 / 8, 256, 0, stream>>>(ao, woutT, b_out, x, out);  // 512 blocks
}
